// MultiHeadedAttention_3607772529124
// MI455X (gfx1250) — compile-verified, matrix operand order checked by reading, not run on hardware
//
#include <hip/hip_runtime.h>
#include <hip/hip_bf16.h>

// ---------------------------------------------------------------------------
// CDNA5 (gfx1250) WMMA / TDM types
// ---------------------------------------------------------------------------
typedef _Float16 h8   __attribute__((ext_vector_type(8)));
typedef _Float16 v16h __attribute__((ext_vector_type(16)));
typedef float    v8f  __attribute__((ext_vector_type(8)));
typedef float    f4   __attribute__((ext_vector_type(4)));
typedef unsigned int u32x4 __attribute__((ext_vector_type(4)));
typedef int      i32x4 __attribute__((ext_vector_type(4)));
typedef int      i32x8 __attribute__((ext_vector_type(8)));

#define WMMA_F32_F16(A, B, C) \
  __builtin_amdgcn_wmma_f32_16x16x32_f16(false, (A), false, (B), (short)0, (C), false, false)

#if defined(__has_builtin)
#if __has_builtin(__builtin_amdgcn_tensor_load_to_lds) && \
    __has_builtin(__builtin_amdgcn_s_wait_tensorcnt)
#define HAVE_TDM 1
#endif
#endif
#ifndef HAVE_TDM
#define HAVE_TDM 0
#endif

__device__ __forceinline__ v16h mkfrag(h8 lo, h8 hi) {
  return __builtin_shufflevector(lo, hi, 0,1,2,3,4,5,6,7,8,9,10,11,12,13,14,15);
}

// 16x32 f16 WMMA operand fragment; rows K-contiguous at `p` (row = lane&15).
// lanes 0-15: K={0..7,16..23}; lanes 16-31: K={8..15,24..31} -> 2x b128 loads.
__device__ __forceinline__ v16h frag_rows(const _Float16* p, int lane) {
  const int hiw = (lane >> 4) & 1;
  h8 a = *(const h8*)(p + hiw * 8);
  h8 b = *(const h8*)(p + hiw * 8 + 16);
  return mkfrag(a, b);
}

#if HAVE_TDM
// Issue one TDM 2-D tile load: 128 rows x 32 f16, row stride rowStrideElems
// (f16 units) in global, hardware-padded in LDS to an 80B pitch (LS=40 f16):
//   pad_interval code 3 = pad after every 16 DWORDs (one 64B row)
//   pad_amount   code 3 = 4 DWORDs (16B) of pad
// Descriptor layout per CDNA5 ISA (groups 0/1; 2D so
// groups 2/3 are zero).
__device__ __forceinline__ void tdm_load_tile(unsigned int ldsByteOff,
                                              unsigned long long gaddr,
                                              unsigned int rowStrideElems) {
  u32x4 g0;
  g0[0] = 1u;                                       // count=1 (valid), user mode
  g0[1] = ldsByteOff;                               // lds_addr [63:32]
  g0[2] = (unsigned int)(gaddr & 0xffffffffull);    // global_addr [95:64]
  g0[3] = (unsigned int)((gaddr >> 32) & 0x01ffffffull) | (2u << 30); // [120:96]+type=2
  i32x8 g1;
  g1[0] = (int)((1u << 16)    // data_size = 1 -> 2 bytes
              | (1u << 20)    // pad_enable
              | (3u << 22)    // pad_interval: every 16 DWORDs
              | (3u << 25));  // pad_amount: 4 DWORDs
  g1[1] = (int)(32u << 16);                 // abar=0 | tensor_dim0[15:0]=32
  g1[2] = (int)(128u << 16);                // tensor_dim0[31:16]=0 | tensor_dim1[15:0]=128
  g1[3] = (int)(32u << 16);                 // tensor_dim1[31:16]=0 | tile_dim0=32
  g1[4] = (int)(128u);                      // tile_dim1=128 | tile_dim2=0
  g1[5] = (int)rowStrideElems;              // tensor_dim0_stride[31:0]
  g1[6] = 0;                                // stride0[47:32] | stride1[15:0]
  g1[7] = 0;                                // stride1[47:16]
  i32x4 z4 = {0, 0, 0, 0};
#if __clang_major__ >= 23
  i32x8 z8 = {0, 0, 0, 0, 0, 0, 0, 0};
  __builtin_amdgcn_tensor_load_to_lds(g0, g1, z4, z4, z8, 0);
#else
  __builtin_amdgcn_tensor_load_to_lds(g0, g1, z4, z4, 0);
#endif
}
#endif  // HAVE_TDM

// ---------------------------------------------------------------------------
// f32 -> f16 conversion pass (one-time; removes cvt VALU + halves GEMM traffic)
// ---------------------------------------------------------------------------
__global__ __launch_bounds__(256)
void cvt_f32_f16_kernel(const float* __restrict__ src, _Float16* __restrict__ dst, int n) {
  const int i = (blockIdx.x * 256 + threadIdx.x) * 8;
  if (i >= n) return;
  f4 a = *(const f4*)(src + i);
  f4 b = *(const f4*)(src + i + 4);
  h8 o;
#pragma unroll
  for (int j = 0; j < 4; ++j) { o[j] = (_Float16)a[j]; o[4 + j] = (_Float16)b[j]; }
  *(h8*)(dst + i) = o;
}

// ---------------------------------------------------------------------------
// All-f16 tiled WMMA GEMM:  out[M,N] = A[M,Kd] @ W[N,Kd]^T + bias[N]
// block = 256 threads (8 waves), tile 128x128, K-step 32.
// TDM path: wave 0 DMAs next tiles (double buffered) while 8 waves run WMMA;
// sync = s_wait_tensorcnt on the issuing wave + workgroup barrier.
// ---------------------------------------------------------------------------
template <typename OT>
__global__ __launch_bounds__(256)
void gemm_wmma_f16_kernel(const _Float16* __restrict__ A, const _Float16* __restrict__ W,
                          const float* __restrict__ bias, OT* __restrict__ out,
                          int M, int N, int Kd) {
  constexpr int LS = 40;                   // LDS row pitch in f16 (80B = 64B row + 16B pad)
  __shared__ _Float16 sA[2][128 * LS];
  __shared__ _Float16 sB[2][128 * LS];

  const int tid = threadIdx.x, lane = tid & 31;
  const int wave = tid >> 5;
  const int waveM = wave & 3;              // 4 M-bands of 32
  const int waveN = wave >> 2;             // 2 N-bands of 64
  const int m0 = blockIdx.x * 128;
  const int n0 = blockIdx.y * 128;

  const v8f vz = {};
  v8f acc[2][4];
#pragma unroll
  for (int i = 0; i < 2; ++i)
#pragma unroll
    for (int j = 0; j < 4; ++j) acc[i][j] = vz;

  const int nK = Kd >> 5;

#if HAVE_TDM
  const unsigned long long aG =
      (unsigned long long)(uintptr_t)A + (unsigned long long)m0 * (unsigned long long)Kd * 2ull;
  const unsigned long long bG =
      (unsigned long long)(uintptr_t)W + (unsigned long long)n0 * (unsigned long long)Kd * 2ull;
  const unsigned int aL0 = (unsigned int)(uintptr_t)(void*)&sA[0][0];
  const unsigned int aL1 = (unsigned int)(uintptr_t)(void*)&sA[1][0];
  const unsigned int bL0 = (unsigned int)(uintptr_t)(void*)&sB[0][0];
  const unsigned int bL1 = (unsigned int)(uintptr_t)(void*)&sB[1][0];

  if (tid < 32) {                          // wave 0 is the DMA-issuing wave
    tdm_load_tile(aL0, aG, (unsigned int)Kd);
    tdm_load_tile(bL0, bG, (unsigned int)Kd);
  }
  for (int i = 0; i < nK; ++i) {
    const int cur = i & 1;
    if (tid < 32) {
      if (i + 1 < nK) {                    // prefetch next K-step into other buffer
        const unsigned long long koff = (unsigned long long)(i + 1) * 64ull; // 32 f16
        tdm_load_tile(cur ? aL0 : aL1, aG + koff, (unsigned int)Kd);
        tdm_load_tile(cur ? bL0 : bL1, bG + koff, (unsigned int)Kd);
        __builtin_amdgcn_s_wait_tensorcnt(2);   // current 2 tiles done, next 2 in flight
      } else {
        __builtin_amdgcn_s_wait_tensorcnt(0);
      }
    }
    __syncthreads();                       // publish TDM-written LDS to all waves

    v16h aF[2];
#pragma unroll
    for (int mt = 0; mt < 2; ++mt)
      aF[mt] = frag_rows(&sA[cur][(waveM * 32 + mt * 16 + (lane & 15)) * LS], lane);
#pragma unroll
    for (int nt = 0; nt < 4; ++nt) {
      v16h bF = frag_rows(&sB[cur][(waveN * 64 + nt * 16 + (lane & 15)) * LS], lane);
#pragma unroll
      for (int mt = 0; mt < 2; ++mt)
        acc[mt][nt] = WMMA_F32_F16(aF[mt], bF, acc[mt][nt]);
    }
    __syncthreads();                       // done reading; buffer reusable for DMA
  }
#else
  // Fallback: cooperative b128 copies, single buffer.
  const int lrow = tid >> 1;
  const int lkb = (tid & 1) * 16;
  for (int ks = 0; ks < Kd; ks += 32) {
    __syncthreads();
    {
      const h8* ga = (const h8*)(A + (size_t)(m0 + lrow) * Kd + ks + lkb);
      _Float16* da = &sA[0][lrow * LS + lkb];
      *(h8*)da = ga[0];
      *(h8*)(da + 8) = ga[1];
      const h8* gb = (const h8*)(W + (size_t)(n0 + lrow) * Kd + ks + lkb);
      _Float16* db = &sB[0][lrow * LS + lkb];
      *(h8*)db = gb[0];
      *(h8*)(db + 8) = gb[1];
    }
    __syncthreads();
    v16h aF[2];
#pragma unroll
    for (int mt = 0; mt < 2; ++mt)
      aF[mt] = frag_rows(&sA[0][(waveM * 32 + mt * 16 + (lane & 15)) * LS], lane);
#pragma unroll
    for (int nt = 0; nt < 4; ++nt) {
      v16h bF = frag_rows(&sB[0][(waveN * 64 + nt * 16 + (lane & 15)) * LS], lane);
#pragma unroll
      for (int mt = 0; mt < 2; ++mt)
        acc[mt][nt] = WMMA_F32_F16(aF[mt], bF, acc[mt][nt]);
    }
  }
#endif

  // D layout: lane holds N = lane&15, M = r + 8*(lane>>4) per 16x16 tile.
  const int hiw = lane >> 4, nl = lane & 15;
#pragma unroll
  for (int mt = 0; mt < 2; ++mt) {
#pragma unroll
    for (int nt = 0; nt < 4; ++nt) {
      const int gm = m0 + waveM * 32 + mt * 16 + hiw * 8;
      const int gn = n0 + waveN * 64 + nt * 16 + nl;
      const float bv = bias[gn];
#pragma unroll
      for (int r = 0; r < 8; ++r)
        out[(size_t)(gm + r) * N + gn] = (OT)(acc[mt][nt][r] + bv);
    }
  }
}

// ---------------------------------------------------------------------------
// Fused frame attention per (video, head):
//   logits[f,t] = K_vh[32x64] @ Q_h^T / 8      (WMMA1, D-layout = [f,t])
//   softmax over f, top-16 mask + renorm       (registers + shfl_xor(16))
//   D-layout of WMMA1 == A-layout of WMMA2 (implicit transpose to [t,f])
//   attn[t,d]  = w[t,32f] @ V_vh[32x64]        (WMMA2)
// block = 256 threads, 8 waves x 64 text columns = T=512.
// ---------------------------------------------------------------------------
__global__ __launch_bounds__(256)
void topk_frame_attn_kernel(const _Float16* __restrict__ q,
                            const _Float16* __restrict__ kbuf,
                            const _Float16* __restrict__ vbuf,
                            _Float16* __restrict__ attn) {
  constexpr int SKS = 72;                // 144B row stride (16B multiple)
  constexpr int SVS = 40;                // 80B row stride  (16B multiple)
  __shared__ _Float16 sK [32 * SKS];     // K_vh  [f][d]
  __shared__ _Float16 sVt[64 * SVS];     // V_vh^T [d][f]

  const int vid = blockIdx.x, hh = blockIdx.y;
  const int tid = threadIdx.x, lane = tid & 31, wave = tid >> 5;

  {
    const int f = tid >> 3;
    const int d = (tid & 7) * 8;
    const size_t base = ((((size_t)vid * 32 + f) * 12 + hh) << 6) + d;
    h8 kv = *(const h8*)(kbuf + base);
    *(h8*)(sK + f * SKS + d) = kv;
    h8 vv = *(const h8*)(vbuf + base);
#pragma unroll
    for (int j = 0; j < 8; ++j) sVt[(d + j) * SVS + f] = vv[j];
  }
  __syncthreads();

  const int hiw = lane >> 4, nl = lane & 15;
  const int tBase = wave * 64;
  const v8f vz = {};
  v16h aFrag[4];

#pragma unroll
  for (int nt = 0; nt < 4; ++nt) {
    v8f accF[2] = {vz, vz};
    const int t = tBase + nt * 16 + nl;
#pragma unroll
    for (int ksp = 0; ksp < 2; ++ksp) {
      const _Float16* qrow = q + ((size_t)t * 12 + hh) * 64 + ksp * 32;
      v16h bF = frag_rows(qrow, lane);
#pragma unroll
      for (int mt = 0; mt < 2; ++mt) {
        v16h aF = frag_rows(sK + (mt * 16 + nl) * SKS + ksp * 32, lane);
        accF[mt] = WMMA_F32_F16(aF, bF, accF[mt]);
      }
    }
    // lane: column t, frames f = 8*hiw + r (tile0) and 16+8*hiw+r (tile1);
    // partner lane (lane^16) holds the complementary 16 frames.
    float l0[8], l1[8];
    float mymax = -1e30f;
#pragma unroll
    for (int r = 0; r < 8; ++r) {
      l0[r] = accF[0][r] * 0.125f;       // 1/sqrt(64)
      l1[r] = accF[1][r] * 0.125f;
      mymax = fmaxf(mymax, fmaxf(l0[r], l1[r]));
    }
    const float cmax = fmaxf(mymax, __shfl_xor(mymax, 16, 32));
    float e0[8], e1[8], s = 0.f;
#pragma unroll
    for (int r = 0; r < 8; ++r) {
      e0[r] = __expf(l0[r] - cmax);
      e1[r] = __expf(l1[r] - cmax);
      s += e0[r] + e1[r];
    }
    const float Z = s + __shfl_xor(s, 16, 32);
    float p0[8], p1[8];
#pragma unroll
    for (int r = 0; r < 8; ++r) {
      p0[r] = __shfl_xor(l0[r], 16, 32);
      p1[r] = __shfl_xor(l1[r], 16, 32);
    }
    // keep top-16 of 32: kept iff #(strictly greater) < 16
    float w0[8], w1[8], skept = 0.f;
#pragma unroll
    for (int i = 0; i < 8; ++i) {
      int c0 = 0, c1 = 0;
#pragma unroll
      for (int j = 0; j < 8; ++j) {
        c0 += (l0[j] > l0[i]) + (l1[j] > l0[i]) + (p0[j] > l0[i]) + (p1[j] > l0[i]);
        c1 += (l0[j] > l1[i]) + (l1[j] > l1[i]) + (p0[j] > l1[i]) + (p1[j] > l1[i]);
      }
      w0[i] = (c0 < 16) ? e0[i] : 0.f;
      w1[i] = (c1 < 16) ? e1[i] : 0.f;
      skept += w0[i] + w1[i];
    }
    const float SKp = skept + __shfl_xor(skept, 16, 32);
    const float inv = 1.f / (SKp + 1e-8f * Z);  // softmax/(masked_sum + 1e-8)
    v16h af;
#pragma unroll
    for (int j = 0; j < 8; ++j) af[j] = (_Float16)(w0[j] * inv);
#pragma unroll
    for (int j = 0; j < 8; ++j) af[8 + j] = (_Float16)(w1[j] * inv);
    aFrag[nt] = af;
  }

  v8f acc2[4][4];
#pragma unroll
  for (int i = 0; i < 4; ++i)
#pragma unroll
    for (int j = 0; j < 4; ++j) acc2[i][j] = vz;
#pragma unroll
  for (int nd = 0; nd < 4; ++nd) {
    v16h bF = frag_rows(sVt + (nd * 16 + nl) * SVS, lane);
#pragma unroll
    for (int mt = 0; mt < 4; ++mt)
      acc2[mt][nd] = WMMA_F32_F16(aFrag[mt], bF, acc2[mt][nd]);
  }
#pragma unroll
  for (int mt = 0; mt < 4; ++mt) {
#pragma unroll
    for (int nd = 0; nd < 4; ++nd) {
      const int t = tBase + mt * 16 + hiw * 8;
      const int d = nd * 16 + nl;
#pragma unroll
      for (int r = 0; r < 8; ++r)
        attn[((size_t)vid * 512 + t + r) * 768 + hh * 64 + d] =
            (_Float16)acc2[mt][nd][r];
    }
  }
}

// ---------------------------------------------------------------------------
// Launch: f32->f16 converts -> 3 projection GEMMs -> fused topk attention ->
// output GEMM. Workspace (f16): text 0.39M | video 6.3M | 4 weights 0.59M ea |
// q 0.39M | k 6.3M | v 6.3M | attn 100.7M  => ~245 MB.
// ---------------------------------------------------------------------------
extern "C" void kernel_launch(void* const* d_in, const int* in_sizes, int n_in,
                              void* d_out, int out_size, void* d_ws, size_t ws_size,
                              hipStream_t stream) {
  const float* text  = (const float*)d_in[0];   // [512,768]
  const float* video = (const float*)d_in[1];   // [256,32,768]
  const float* Wq = (const float*)d_in[2];
  const float* bq = (const float*)d_in[3];
  const float* Wk = (const float*)d_in[4];
  const float* bk = (const float*)d_in[5];
  const float* Wv = (const float*)d_in[6];
  const float* bv = (const float*)d_in[7];
  const float* Wo = (const float*)d_in[8];
  const float* bo = (const float*)d_in[9];
  float* out = (float*)d_out;                   // [256,512,768]

  _Float16* p = (_Float16*)d_ws;
  _Float16* text16  = p; p += (size_t)512 * 768;
  _Float16* video16 = p; p += (size_t)8192 * 768;
  _Float16* wq16 = p; p += (size_t)768 * 768;
  _Float16* wk16 = p; p += (size_t)768 * 768;
  _Float16* wv16 = p; p += (size_t)768 * 768;
  _Float16* wo16 = p; p += (size_t)768 * 768;
  _Float16* q    = p; p += (size_t)512 * 768;
  _Float16* kbuf = p; p += (size_t)8192 * 768;
  _Float16* vbuf = p; p += (size_t)8192 * 768;
  _Float16* attn = p;

  dim3 blk(256);
  // one-time f32 -> f16 conversions
  cvt_f32_f16_kernel<<<192,  blk, 0, stream>>>(text,  text16,  512 * 768);
  cvt_f32_f16_kernel<<<3072, blk, 0, stream>>>(video, video16, 8192 * 768);
  cvt_f32_f16_kernel<<<288,  blk, 0, stream>>>(Wq, wq16, 768 * 768);
  cvt_f32_f16_kernel<<<288,  blk, 0, stream>>>(Wk, wk16, 768 * 768);
  cvt_f32_f16_kernel<<<288,  blk, 0, stream>>>(Wv, wv16, 768 * 768);
  cvt_f32_f16_kernel<<<288,  blk, 0, stream>>>(Wo, wo16, 768 * 768);
  // q/k/v projections (f16 in -> f16 out), TDM double-buffered tiles
  gemm_wmma_f16_kernel<_Float16><<<dim3(4, 6),   blk, 0, stream>>>(text16,  wq16, bq, q,    512,  768, 768);
  gemm_wmma_f16_kernel<_Float16><<<dim3(64, 6),  blk, 0, stream>>>(video16, wk16, bk, kbuf, 8192, 768, 768);
  gemm_wmma_f16_kernel<_Float16><<<dim3(64, 6),  blk, 0, stream>>>(video16, wv16, bv, vbuf, 8192, 768, 768);
  // fused per-(video,head) frame attention with top-16 mask
  topk_frame_attn_kernel<<<dim3(256, 12), blk, 0, stream>>>(q, kbuf, vbuf, attn);
  // output projection (f16 in -> f32 out), the 154 GF GEMM
  gemm_wmma_f16_kernel<float><<<dim3(1024, 6), blk, 0, stream>>>(attn, wo16, bo, out, 131072, 768, 768);
}
